// GraphAttentionLayer_38843684225825
// MI455X (gfx1250) — matrix operand order repaired and checked by reading, compiled, not run on hardware
//
#include <hip/hip_runtime.h>

typedef __attribute__((ext_vector_type(16))) __bf16 v16bf;
typedef __attribute__((ext_vector_type(8)))  float  v8f;

union BF16x16 { v16bf v; unsigned u[8]; };

#define NN 8192
#define FIN 512
#define FOUT 64
#define ALPHA 0.2f

// fp32 -> bf16 via native convert (backend picks v_cvt_pk_bf16_f32 where possible)
__device__ __forceinline__ unsigned bf16pack(float lo, float hi) {
    unsigned short a = __builtin_bit_cast(unsigned short, (__bf16)lo);
    unsigned short b = __builtin_bit_cast(unsigned short, (__bf16)hi);
    return (unsigned)a | ((unsigned)b << 16);
}
__device__ __forceinline__ unsigned short bf16one(float x) {
    return __builtin_bit_cast(unsigned short, (__bf16)x);
}

// unnormalized masked softmax weight
__device__ __forceinline__ float pval(float w1, float w2, int ad, float em) {
    float e = w1 + w2;
    e = e > 0.f ? e : ALPHA * e;
    return ad > 0 ? __expf(e - em) : 0.f;
}

// ---------------------------------------------------------------------------
// Kernel 1: Wh = h @ W via v_wmma_f32_16x16x32_bf16.
// grid 512 (16-row tiles), block 128 = 4 waves, wave w -> output cols [16w,16w+16)
// ---------------------------------------------------------------------------
__global__ void wh_kernel(const float* __restrict__ h, const float* __restrict__ W,
                          float* __restrict__ Wh, unsigned short* __restrict__ WhT) {
    const int lane = threadIdx.x & 31;
    const int wid  = threadIdx.x >> 5;
    const int r0   = blockIdx.x * 16;
    const int c0   = wid * 16;
    const int m    = lane & 15;   // A: M row / B,C: N col
    const int hh   = lane >> 4;   // lane half

    v8f acc = {};
    for (int kb = 0; kb < FIN; kb += 32) {
        // A tile (16x32 bf16): K runs [kb+16h, +16)
        const float* hp = h + (r0 + m) * FIN + kb + 16 * hh;
        float4 a0 = *(const float4*)(hp);
        float4 a1 = *(const float4*)(hp + 4);
        float4 a2 = *(const float4*)(hp + 8);
        float4 a3 = *(const float4*)(hp + 12);
        BF16x16 A;
        A.v = (v16bf){(__bf16)a0.x, (__bf16)a0.y, (__bf16)a0.z, (__bf16)a0.w,
                      (__bf16)a1.x, (__bf16)a1.y, (__bf16)a1.z, (__bf16)a1.w,
                      (__bf16)a2.x, (__bf16)a2.y, (__bf16)a2.z, (__bf16)a2.w,
                      (__bf16)a3.x, (__bf16)a3.y, (__bf16)a3.z, (__bf16)a3.w};
        // B tile (32x16 bf16): lane col = m, VGPR v holds K = 2v + 16h (pair)
        BF16x16 B;
        const float* wp = W + (kb + 16 * hh) * FOUT + c0 + m;
        #pragma unroll
        for (int v = 0; v < 8; ++v)
            B.u[v] = bf16pack(wp[(2 * v) * FOUT], wp[(2 * v + 1) * FOUT]);
        acc = __builtin_amdgcn_wmma_f32_16x16x32_bf16(false, A.v, false, B.v,
                                                      (short)0, acc, false, false);
    }
    // C layout: VGPR r -> M = r + 8h, N = lane&15
    #pragma unroll
    for (int r = 0; r < 8; ++r) {
        int M = r + 8 * hh;
        float val = acc[r];
        Wh[(r0 + M) * FOUT + c0 + m] = val;                 // fp32 for a-projection
        WhT[(c0 + m) * NN + (r0 + M)] = bf16one(val);       // bf16 transposed for attn GEMM
    }
}

// ---------------------------------------------------------------------------
// Kernel 2: Wh1[i] = Wh[i,:] . a[:64],  Wh2[i] = Wh[i,:] . a[64:]
// ---------------------------------------------------------------------------
__global__ void attn_vec_kernel(const float* __restrict__ Wh, const float* __restrict__ a,
                                float* __restrict__ Wh1, float* __restrict__ Wh2) {
    int i = blockIdx.x * blockDim.x + threadIdx.x;
    if (i >= NN) return;
    const float* row = Wh + i * FOUT;
    float s1 = 0.f, s2 = 0.f;
    #pragma unroll 8
    for (int c = 0; c < FOUT; ++c) {
        float v = row[c];
        s1 += v * a[c];
        s2 += v * a[FOUT + c];
    }
    Wh1[i] = s1;
    Wh2[i] = s2;
}

// ---------------------------------------------------------------------------
// Kernel 3: emax[i] = leaky_relu(Wh1[i] + max_{adj[i,j]>0} Wh2[j])  (adj pass #1)
// ---------------------------------------------------------------------------
__global__ void rowmax_kernel(const int* __restrict__ adj, const float* __restrict__ Wh1,
                              const float* __restrict__ Wh2, float* __restrict__ emax) {
    __shared__ float red[256];
    const int i = blockIdx.x;
    const int* row = adj + (size_t)i * NN;
    float mx = -__builtin_inff();
    for (int j = threadIdx.x; j < NN; j += 256) {
        float w = Wh2[j];
        mx = fmaxf(mx, row[j] > 0 ? w : -__builtin_inff());
    }
    red[threadIdx.x] = mx;
    __syncthreads();
    for (int s = 128; s > 0; s >>= 1) {
        if (threadIdx.x < s) red[threadIdx.x] = fmaxf(red[threadIdx.x], red[threadIdx.x + s]);
        __syncthreads();
    }
    if (threadIdx.x == 0) {
        float mm = red[0];
        float e = 0.f;
        if (mm != -__builtin_inff()) {
            float x = Wh1[i] + mm;
            e = x > 0.f ? x : ALPHA * x;
        }
        emax[i] = e;
    }
}

// ---------------------------------------------------------------------------
// Kernel 4: fused masked-softmax x Wh via WMMA (adj pass #2).
// grid 512 (16-row tiles), block 128 = 4 waves splitting the j-loop.
// ---------------------------------------------------------------------------
__global__ void attn_kernel(const int* __restrict__ adj,
                            const unsigned short* __restrict__ WhT,
                            const float* __restrict__ Wh1,
                            const float* __restrict__ Wh2,
                            const float* __restrict__ emax,
                            float* __restrict__ out) {
    __shared__ float lds_acc[16 * FOUT];
    __shared__ float lds_sum[16];

    const int lane = threadIdx.x & 31;
    const int wid  = threadIdx.x >> 5;
    const int i0   = blockIdx.x * 16;
    const int m    = lane & 15;
    const int hh   = lane >> 4;

    for (int t = threadIdx.x; t < 16 * FOUT; t += 128) lds_acc[t] = 0.f;
    if (threadIdx.x < 16) lds_sum[threadIdx.x] = 0.f;
    __syncthreads();

    const float w1 = Wh1[i0 + m];
    const float em = emax[i0 + m];
    const int*  adjrow = adj + (size_t)(i0 + m) * NN;

    v8f acc0 = {}, acc1 = {}, acc2 = {}, acc3 = {};
    float psum = 0.f;

    const int jbeg = wid * (NN / 4);
    const int jend = jbeg + (NN / 4);
    for (int jb = jbeg; jb < jend; jb += 32) {
        const int b1 = jb + 16 * hh;
        const int b2 = jb + 8 + 16 * hh;
        // prefetch next chunk's adj segment (dominant HBM stream)
        if (jb + 32 < jend) {
            __builtin_prefetch(adjrow + b1 + 32, 0, 1);
            __builtin_prefetch(adjrow + b2 + 32, 0, 1);
        }
        float4 w2a = *(const float4*)(Wh2 + b1);
        float4 w2b = *(const float4*)(Wh2 + b1 + 4);
        float4 w2c = *(const float4*)(Wh2 + b2);
        float4 w2d = *(const float4*)(Wh2 + b2 + 4);
        int4 ada = *(const int4*)(adjrow + b1);
        int4 adb = *(const int4*)(adjrow + b1 + 4);
        int4 adc = *(const int4*)(adjrow + b2);
        int4 add = *(const int4*)(adjrow + b2 + 4);

        float p0  = pval(w1, w2a.x, ada.x, em), p1  = pval(w1, w2a.y, ada.y, em);
        float p2  = pval(w1, w2a.z, ada.z, em), p3  = pval(w1, w2a.w, ada.w, em);
        float p4  = pval(w1, w2b.x, adb.x, em), p5  = pval(w1, w2b.y, adb.y, em);
        float p6  = pval(w1, w2b.z, adb.z, em), p7  = pval(w1, w2b.w, adb.w, em);
        float p8  = pval(w1, w2c.x, adc.x, em), p9  = pval(w1, w2c.y, adc.y, em);
        float p10 = pval(w1, w2c.z, adc.z, em), p11 = pval(w1, w2c.w, adc.w, em);
        float p12 = pval(w1, w2d.x, add.x, em), p13 = pval(w1, w2d.y, add.y, em);
        float p14 = pval(w1, w2d.z, add.z, em), p15 = pval(w1, w2d.w, add.w, em);

        psum += ((p0 + p1) + (p2 + p3)) + ((p4 + p5) + (p6 + p7))
              + ((p8 + p9) + (p10 + p11)) + ((p12 + p13) + (p14 + p15));

        BF16x16 A;
        A.v = (v16bf){(__bf16)p0,  (__bf16)p1,  (__bf16)p2,  (__bf16)p3,
                      (__bf16)p4,  (__bf16)p5,  (__bf16)p6,  (__bf16)p7,
                      (__bf16)p8,  (__bf16)p9,  (__bf16)p10, (__bf16)p11,
                      (__bf16)p12, (__bf16)p13, (__bf16)p14, (__bf16)p15};

        // B tiles: WhT[(c0+m)*NN + jb + 16h .. +15] -> v=0..7 (K pairs), contiguous b128s
        #pragma unroll
        for (int c = 0; c < 4; ++c) {
            const unsigned short* bp = WhT + (c * 16 + m) * NN + jb + 16 * hh;
            uint4 q0 = *(const uint4*)(bp);
            uint4 q1 = *(const uint4*)(bp + 8);
            BF16x16 B;
            B.u[0] = q0.x; B.u[1] = q0.y; B.u[2] = q0.z; B.u[3] = q0.w;
            B.u[4] = q1.x; B.u[5] = q1.y; B.u[6] = q1.z; B.u[7] = q1.w;
            if (c == 0) acc0 = __builtin_amdgcn_wmma_f32_16x16x32_bf16(false, A.v, false, B.v, (short)0, acc0, false, false);
            if (c == 1) acc1 = __builtin_amdgcn_wmma_f32_16x16x32_bf16(false, A.v, false, B.v, (short)0, acc1, false, false);
            if (c == 2) acc2 = __builtin_amdgcn_wmma_f32_16x16x32_bf16(false, A.v, false, B.v, (short)0, acc2, false, false);
            if (c == 3) acc3 = __builtin_amdgcn_wmma_f32_16x16x32_bf16(false, A.v, false, B.v, (short)0, acc3, false, false);
        }
    }

    // merge per-wave partial tiles (linear: global emax already applied)
    #pragma unroll
    for (int r = 0; r < 8; ++r) {
        int M = r + 8 * hh;
        atomicAdd(&lds_acc[M * FOUT +  0 + m], acc0[r]);
        atomicAdd(&lds_acc[M * FOUT + 16 + m], acc1[r]);
        atomicAdd(&lds_acc[M * FOUT + 32 + m], acc2[r]);
        atomicAdd(&lds_acc[M * FOUT + 48 + m], acc3[r]);
    }
    atomicAdd(&lds_sum[m], psum);
    __syncthreads();

    for (int t = threadIdx.x; t < 16 * FOUT; t += 128) {
        int row = t >> 6, col = t & 63;
        float s = lds_sum[row];
        s = s > 1e-30f ? s : 1e-30f;
        float v = lds_acc[t] / s;
        v = v > 0.f ? v : (__expf(v) - 1.f);   // ELU
        out[(i0 + row) * FOUT + col] = v;
    }
}

extern "C" void kernel_launch(void* const* d_in, const int* in_sizes, int n_in,
                              void* d_out, int out_size, void* d_ws, size_t ws_size,
                              hipStream_t stream) {
    const float* h   = (const float*)d_in[0];
    const int*   adj = (const int*)d_in[1];
    const float* W   = (const float*)d_in[2];
    const float* a   = (const float*)d_in[3];
    float* out = (float*)d_out;

    char* ws = (char*)d_ws;
    float*          Wh   = (float*)ws;                                    // 8192*64*4  = 2 MB
    unsigned short* WhT  = (unsigned short*)(ws + (size_t)NN * FOUT * 4); // 64*8192*2 = 1 MB
    float*          Wh1  = (float*)(ws + 3u * 1024u * 1024u);             // 32 KB
    float*          Wh2  = Wh1 + NN;                                      // 32 KB
    float*          emax = Wh2 + NN;                                      // 32 KB

    wh_kernel      <<<NN / 16, 128, 0, stream>>>(h, W, Wh, WhT);
    attn_vec_kernel<<<NN / 256, 256, 0, stream>>>(Wh, a, Wh1, Wh2);
    rowmax_kernel  <<<NN, 256, 0, stream>>>(adj, Wh1, Wh2, emax);
    attn_kernel    <<<NN / 16, 128, 0, stream>>>(adj, WhT, Wh1, Wh2, emax, out);
}
